// DyDCNv2_11802570129664
// MI455X (gfx1250) — matrix operand order repaired and checked by reading, not compiled, not run on hardware
//
#include <hip/hip_runtime.h>
#include <hip/hip_bf16.h>

// ---------------------------------------------------------------------------
// DCNv2 (3x3, s1, p1) + GroupNorm(16) for MI455X (gfx1250, wave32, WMMA).
// Implicit GEMM  out[b, o, p] = sum_{ck} W[o, ck] * val[ck, p],
//   ck = k_pt*256 + c  (K=32 WMMA chunk = 1 tap x 32 channels),
//   val = bilinear-gathered, mask-modulated x (NHWC for coalescing),
//   v_wmma_f32_16x16x32_bf16, A staged via global_load_async_to_lds_b128
//   (double-buffered, overlapped with the gather), B built in LDS.
// ---------------------------------------------------------------------------

#define Bn 8
#define Cn 256
#define On 256
#define Hn 64
#define Wn 64
#define HWn 4096
#define KPTS 9
#define KSTEPS 72          // 2304 / 32
#define NTILE 64           // pixels per block (8x8 spatial)
#define NBLOCKS 512        // Bn * (HWn / NTILE)
#define ASTEP_BYTES 16384  // 16 A-fragments (16x32 bf16 each) per K-step

typedef __attribute__((ext_vector_type(16))) __bf16 bf16x16;
typedef __attribute__((ext_vector_type(8)))  float  f32x8;

__device__ __forceinline__ unsigned short f2bf(float f) {
    unsigned int u = __float_as_uint(f);
    u += 0x7fffu + ((u >> 16) & 1u);   // round-to-nearest-even
    return (unsigned short)(u >> 16);
}

// --------------------------- x: NCHW -> NHWC -------------------------------
__global__ __launch_bounds__(256) void transpose_x(const float* __restrict__ x,
                                                   float* __restrict__ xt) {
    __shared__ float tsh[32][33];
    int bId = blockIdx.x;              // b * 1024 + pt * 8 + ct
    int ct = bId & 7;
    int pt = (bId >> 3) & 127;
    int b  = bId >> 10;
    int c0 = ct << 5, p0 = pt << 5;
    int tx = threadIdx.x & 31, ty = threadIdx.x >> 5;
#pragma unroll
    for (int i = 0; i < 4; ++i) {
        int cl = ty + i * 8;
        tsh[cl][tx] = x[(((size_t)b * Cn + c0 + cl) << 12) + p0 + tx];
    }
    __syncthreads();
#pragma unroll
    for (int i = 0; i < 4; ++i) {
        int pl = ty + i * 8;
        xt[(((size_t)b * HWn + p0 + pl) << 8) + c0 + tx] = tsh[tx][pl];
    }
}

// ---------------- weight: f32 [O,C,3,3] -> bf16 A-fragments ----------------
// Fragment f = kstep*16 + m_sub.  Lane L holds M = m_sub*16 + (L&15),
// half j holds K = kstep*32 + (j>>3)*16 + ((L>>4)&1)*8 + (j&7)   (ISA layout).
__global__ __launch_bounds__(256) void convert_w(const float* __restrict__ w,
                                                 unsigned short* __restrict__ wb) {
    int t = blockIdx.x * 256 + threadIdx.x;
    int frag = t >> 5;
    int lane = t & 31;
    if (frag >= KSTEPS * 16) return;
    int kstep = frag >> 4;
    int m_sub = frag & 15;
    int o = m_sub * 16 + (lane & 15);
    unsigned short vals[16];
#pragma unroll
    for (int j = 0; j < 16; ++j) {
        int Kloc = ((j >> 3) << 4) + ((lane >> 4) << 3) + (j & 7);
        int K = kstep * 32 + Kloc;     // ck = kpt*256 + c
        int kpt = K >> 8;
        int c = K & 255;
        vals[j] = f2bf(w[((size_t)o * Cn + c) * 9 + kpt]);
    }
    size_t base = ((size_t)frag * 32 + lane) * 16;
#pragma unroll
    for (int j = 0; j < 16; ++j) wb[base + j] = vals[j];
}

// ------------------------------- main kernel -------------------------------
__global__ __launch_bounds__(256) void dcn_main(const float* __restrict__ xt,
                                                const float* __restrict__ offset,
                                                const float* __restrict__ mask,
                                                const unsigned short* __restrict__ wb,
                                                float* __restrict__ out,
                                                float* __restrict__ stats) {
    __shared__ int   sIdx[KPTS * NTILE * 4];                  //  9216 B
    __shared__ float sWgt[KPTS * NTILE * 4];                  //  9216 B
    __shared__ __align__(32) unsigned int   sB32[4 * 32 * 8]; //  4096 B (B frags)
    __shared__ __align__(32) unsigned short sA[2][ASTEP_BYTES / 2]; // 32 KB

    const int tid  = threadIdx.x;
    const int lane = tid & 31;
    const int wave = tid >> 5;

    const int tile = blockIdx.x;       // 0..511
    const int b    = tile >> 6;
    const int t64  = tile & 63;
    const int h0   = (t64 >> 3) << 3;
    const int w0   = (t64 & 7) << 3;

    const unsigned sA_base = (unsigned)(size_t)(void*)&sA[0][0];
    const unsigned long long wb_u64 = (unsigned long long)(const void*)wb;

    // ---- prologue: kick off async copy of A-fragments for K-step 0 ----
    {
        unsigned l = sA_base + (unsigned)(tid * 64);
        unsigned v = (unsigned)(tid * 64);
#pragma unroll
        for (int q = 0; q < 4; ++q) {
            asm volatile("global_load_async_to_lds_b128 %0, %1, %2"
                         :: "v"(l + q * 16), "v"(v + q * 16), "s"(wb_u64)
                         : "memory");
        }
    }

    // ---- Stage 0: bilinear metadata for 9 taps x 64 pixels ----
    for (int item = tid; item < KPTS * NTILE; item += 256) {
        int kpt = item >> 6;
        int n   = item & 63;
        int h = h0 + (n >> 3), w = w0 + (n & 7);
        int hw = (h << 6) + w;
        float dy = offset[(((size_t)b * 18 + 2 * kpt    ) << 12) + hw];
        float dx = offset[(((size_t)b * 18 + 2 * kpt + 1) << 12) + hw];
        float m  = mask  [(((size_t)b * 9  + kpt        ) << 12) + hw];
        float py = (float)h + (float)(kpt / 3 - 1) + dy;
        float px = (float)w + (float)(kpt % 3 - 1) + dx;
        float y0f = floorf(py), x0f = floorf(px);
        float wy = py - y0f, wx = px - x0f;
        int y0 = (int)y0f, x0i = (int)x0f;
        float cw[4] = {(1.f - wy) * (1.f - wx), (1.f - wy) * wx,
                       wy * (1.f - wx),          wy * wx};
        int yy[4] = {y0, y0, y0 + 1, y0 + 1};
        int xx[4] = {x0i, x0i + 1, x0i, x0i + 1};
#pragma unroll
        for (int c4 = 0; c4 < 4; ++c4) {
            bool valid = (yy[c4] >= 0) && (yy[c4] <= Hn - 1) &&
                         (xx[c4] >= 0) && (xx[c4] <= Wn - 1);
            int yi = min(max(yy[c4], 0), Hn - 1);
            int xi = min(max(xx[c4], 0), Wn - 1);
            sIdx[item * 4 + c4] = (yi << 6) + xi;
            sWgt[item * 4 + c4] = valid ? cw[c4] * m : 0.0f;
        }
    }
    __syncthreads();   // meta visible; also orders prologue vs. loop barriers

    const int n_sub  = wave & 3;       // which N 16-subtile (0..3)
    const int m_half = wave >> 2;      // 0/1 -> M subtiles [mh*8, mh*8+8)
    f32x8 acc[8];
#pragma unroll
    for (int i = 0; i < 8; ++i) acc[i] = (f32x8)(0.0f);

    // build-phase mapping: thread -> channel-pair km (0..15) x 4 pixels
    const int km  = tid & 15;          // K = 2*km, 2*km+1 within the chunk
    const int grp = tid >> 4;          // pixel base; n = grp + 16*j
    const float* xtb = xt + ((size_t)b << 20);          // b * HW * C
    const unsigned short* sBh = (const unsigned short*)sB32;

    for (int s = 0; s < KSTEPS; ++s) {
        const int cur = s & 1, nxt = cur ^ 1;
        const int kpt = s / 8;
        const int c0  = (s & 7) << 5;

        // ---- async prefetch of next K-step's A-fragments into sA[nxt] ----
        if (s + 1 < KSTEPS) {
            unsigned l = sA_base + (unsigned)(nxt * ASTEP_BYTES + tid * 64);
            unsigned v = (unsigned)(tid * 64);
            unsigned long long g = wb_u64 + (unsigned long long)(s + 1) * ASTEP_BYTES;
#pragma unroll
            for (int q = 0; q < 4; ++q) {
                asm volatile("global_load_async_to_lds_b128 %0, %1, %2"
                             :: "v"(l + q * 16), "v"(v + q * 16), "s"(g)
                             : "memory");
            }
        }

        // ---- build B tile (32 ch x 64 px) as packed bf16 pairs ----
#pragma unroll
        for (int j = 0; j < 4; ++j) {
            int n  = grp + (j << 4);
            int mi = (((kpt << 6) + n) << 2);
            float vx = 0.f, vy = 0.f;
#pragma unroll
            for (int c4 = 0; c4 < 4; ++c4) {
                float wv = sWgt[mi + c4];
                int idx  = sIdx[mi + c4];
                const float2 p = *reinterpret_cast<const float2*>(
                    xtb + ((size_t)idx << 8) + c0 + (km << 1));
                vx += wv * p.x;
                vy += wv * p.y;
            }
            unsigned packed = (unsigned)f2bf(vx) | ((unsigned)f2bf(vy) << 16);
            int fraglane = (((km >> 2) & 1) << 4) + (n & 15);
            sB32[((((n >> 4) << 5) + fraglane) << 3) + ((km & 3) | ((km >> 3) << 2))] = packed;
        }

        // previous batch of async A-copies must have landed (in-order retire)
        if (s + 1 < KSTEPS) asm volatile("s_wait_asynccnt 0x4" ::: "memory");
        else                asm volatile("s_wait_asynccnt 0x0" ::: "memory");
        __syncthreads();   // sB built + sA[cur] complete, visible to all waves

        // ---- WMMA: 8 x (16x16) tiles per wave, A from LDS ----
        bf16x16 bfrag =
            *reinterpret_cast<const bf16x16*>(&sBh[((n_sub << 5) + lane) << 4]);
#pragma unroll
        for (int i = 0; i < 8; ++i) {
            const int m_sub = m_half * 8 + i;
            const bf16x16 a = *reinterpret_cast<const bf16x16*>(
                &sA[cur][((m_sub << 5) + lane) << 4]);
            acc[i] = __builtin_amdgcn_wmma_f32_16x16x32_bf16(
                false, a, false, bfrag, (short)0, acc[i], false, false);
        }
        __syncthreads();   // everyone consumed sB / sA[cur]
    }

    // ---- epilogue: store conv result + per-(b,group) sum/sumsq ----
    const int nloc = (n_sub << 4) + (lane & 15);
    const int h = h0 + (nloc >> 3), w = w0 + (nloc & 7);
    const int hw = (h << 6) + w;
    const int mrow = (lane >> 4) << 3;   // 0 or 8 (C/D VGPR layout)
#pragma unroll
    for (int i = 0; i < 8; ++i) {
        const int m_sub = m_half * 8 + i;   // == GroupNorm group (O/16 == 16)
        float s = 0.f, s2 = 0.f;
#pragma unroll
        for (int r = 0; r < 8; ++r) {
            float v = acc[i][r];
            int o = (m_sub << 4) + mrow + r;
            out[(((size_t)b * On + o) << 12) + hw] = v;
            s += v; s2 += v * v;
        }
#pragma unroll
        for (int off = 16; off > 0; off >>= 1) {
            s  += __shfl_xor(s,  off, 32);
            s2 += __shfl_xor(s2, off, 32);
        }
        if (lane == 0) {
            atomicAdd(&stats[(b * 16 + m_sub) * 2 + 0], s);
            atomicAdd(&stats[(b * 16 + m_sub) * 2 + 1], s2);
        }
    }
}

// --------------------------- GroupNorm apply -------------------------------
__global__ __launch_bounds__(256) void gn_apply(float* __restrict__ out,
                                                const float* __restrict__ stats,
                                                const float* __restrict__ gamma,
                                                const float* __restrict__ beta) {
    size_t t = (size_t)blockIdx.x * 256 + threadIdx.x;
    size_t base = t * 4;
    int bo = (int)(base >> 12);
    int o = bo & 255;
    int b = bo >> 8;
    int g = o >> 4;
    float s  = stats[(b * 16 + g) * 2 + 0];
    float s2 = stats[(b * 16 + g) * 2 + 1];
    const float inv = 1.0f / 65536.0f;           // 16 ch * 4096 px
    float mean = s * inv;
    float var  = s2 * inv - mean * mean;
    float rstd = rsqrtf(var + 1e-5f);
    float ga = gamma[o], be = beta[o];
    float4 v = *reinterpret_cast<float4*>(out + base);
    v.x = (v.x - mean) * rstd * ga + be;
    v.y = (v.y - mean) * rstd * ga + be;
    v.z = (v.z - mean) * rstd * ga + be;
    v.w = (v.w - mean) * rstd * ga + be;
    *reinterpret_cast<float4*>(out + base) = v;
}

// ------------------------------- launcher ----------------------------------
extern "C" void kernel_launch(void* const* d_in, const int* in_sizes, int n_in,
                              void* d_out, int out_size, void* d_ws, size_t ws_size,
                              hipStream_t stream) {
    const float* x      = (const float*)d_in[0];
    const float* offset = (const float*)d_in[1];
    const float* mask   = (const float*)d_in[2];
    const float* weight = (const float*)d_in[3];
    const float* gamma  = (const float*)d_in[4];
    const float* beta   = (const float*)d_in[5];
    float* out = (float*)d_out;

    const size_t XT_BYTES = (size_t)Bn * Cn * HWn * sizeof(float);      // 33.5 MB
    const size_t WB_BYTES = (size_t)KSTEPS * ASTEP_BYTES;               // 1.15 MB
    char* ws = (char*)d_ws;
    float*          xt    = (float*)ws;
    unsigned short* wb    = (unsigned short*)(ws + XT_BYTES);
    float*          stats = (float*)(ws + XT_BYTES + WB_BYTES);

    hipMemsetAsync(stats, 0, Bn * 16 * 2 * sizeof(float), stream);
    transpose_x<<<8192, 256, 0, stream>>>(x, xt);
    convert_w<<<(KSTEPS * 16 * 32 + 255) / 256, 256, 0, stream>>>(weight, wb);
    dcn_main<<<NBLOCKS, 256, 0, stream>>>(xt, offset, mask, wb, out, stats);
    gn_apply<<<(Bn * On * HWn / 4 + 255) / 256, 256, 0, stream>>>(out, stats, gamma, beta);
}
